// GraphEncoder_17935783428096
// MI455X (gfx1250) — matrix operand order repaired and checked by reading, not compiled, not run on hardware
//
#include <hip/hip_runtime.h>
#include <hip/hip_fp16.h>

#define NN     100000      // nodes
#define NE     1600000     // edges
#define NH     4           // heads
#define HC     64          // channels per head
#define FDIM   256         // NH*HC
#define NG     64          // graphs
#define LRELU  0.2f

typedef __attribute__((ext_vector_type(16))) _Float16 v16h;
typedef __attribute__((ext_vector_type(8)))  _Float16 v8h;
typedef __attribute__((ext_vector_type(8)))  float    v8f;

// ---------------- helpers ----------------

__device__ __forceinline__ float leaky(float v) {
  return v > 0.0f ? v : LRELU * v;
}

// float atomic max via sign-split int/uint ordering trick (init must be -inf)
__device__ __forceinline__ void atomicMaxF(float* addr, float val) {
  if (val >= 0.0f) atomicMax((int*)addr, __float_as_int(val));
  else             atomicMin((unsigned int*)addr, __float_as_uint(val));
}

// ---------------- small utility kernels ----------------

__global__ void k_zero(float* p, int n) {
  int i = blockIdx.x * blockDim.x + threadIdx.x;
  if (i < n) p[i] = 0.0f;
}

__global__ void k_cvt_f16(const float* __restrict__ src, _Float16* __restrict__ dst, int n) {
  int i = blockIdx.x * blockDim.x + threadIdx.x;
  if (i < n) dst[i] = (_Float16)src[i];
}

// ---------------- layer 1 transform: xl = x @ W1   (K=3, scalar) ----------------

__global__ void k_l1_transform(const float* __restrict__ x, const float* __restrict__ W1,
                               float* __restrict__ xl) {
  int t = blockIdx.x * blockDim.x + threadIdx.x;   // NN*64 threads, 4 cols each
  if (t >= NN * 64) return;
  int n  = t >> 6;
  int j4 = (t & 63) << 2;
  float x0 = x[n * 3 + 0], x1 = x[n * 3 + 1], x2 = x[n * 3 + 2];
#pragma unroll
  for (int u = 0; u < 4; ++u) {
    int j = j4 + u;
    xl[n * FDIM + j] = fmaf(x0, W1[j], fmaf(x1, W1[FDIM + j], x2 * W1[2 * FDIM + j]));
  }
}

// ---------------- per-node attention logits: es/ed = <xl[n,h,:], a> ----------------

__global__ void k_attn_logits(const float* __restrict__ xl, const float* __restrict__ a_src,
                              const float* __restrict__ a_dst,
                              float* __restrict__ es, float* __restrict__ ed) {
  int t = blockIdx.x * blockDim.x + threadIdx.x;   // NN*NH
  if (t >= NN * NH) return;
  int n = t >> 2, h = t & 3;
  const float4* r4 = (const float4*)(xl + n * FDIM + h * HC);
  const float4* s4 = (const float4*)(a_src + h * HC);
  const float4* d4 = (const float4*)(a_dst + h * HC);
  float s0 = 0.0f, s1 = 0.0f;
#pragma unroll 4
  for (int i = 0; i < HC / 4; ++i) {
    float4 r = r4[i], a = s4[i], b = d4[i];
    s0 += r.x * a.x + r.y * a.y + r.z * a.z + r.w * a.w;
    s1 += r.x * b.x + r.y * b.y + r.z * b.z + r.w * b.w;
  }
  es[t] = s0;
  ed[t] = s1;
}

// ---------------- per-node init: m=-inf, s=0, acc=bias ----------------

__global__ void k_node_init(float* __restrict__ m, float* __restrict__ s,
                            float* __restrict__ acc, const float* __restrict__ b) {
  int t = blockIdx.x * blockDim.x + threadIdx.x;   // NN*64
  if (t >= NN * 64) return;
  int n = t >> 6, c = t & 63;
  acc[t] = b[c];
  if (c < NH) {
    m[n * NH + c] = __int_as_float(0xFF800000u);   // -inf
    s[n * NH + c] = 0.0f;
  }
}

// ---------------- edge pass 1: segment max ----------------

__global__ void k_edge_max(const int* __restrict__ src, const int* __restrict__ dst,
                           const float* __restrict__ es, const float* __restrict__ ed,
                           float* __restrict__ m) {
  int e = blockIdx.x * blockDim.x + threadIdx.x;
  if (e >= NE) return;
  int u = src[e], v = dst[e];
#pragma unroll
  for (int h = 0; h < NH; ++h) {
    float val = leaky(es[u * NH + h] + ed[v * NH + h]);
    atomicMaxF(&m[v * NH + h], val);
  }
}

// ---------------- edge pass 2: exp + segment sum ----------------

__global__ void k_edge_expsum(const int* __restrict__ src, const int* __restrict__ dst,
                              const float* __restrict__ es, const float* __restrict__ ed,
                              const float* __restrict__ m,
                              float* __restrict__ exbuf, float* __restrict__ s) {
  int e = blockIdx.x * blockDim.x + threadIdx.x;
  if (e >= NE) return;
  int u = src[e], v = dst[e];
#pragma unroll
  for (int h = 0; h < NH; ++h) {
    float val = leaky(es[u * NH + h] + ed[v * NH + h]);
    float ex  = __expf(val - m[v * NH + h]);
    exbuf[e * NH + h] = ex;
    atomicAdd(&s[v * NH + h], ex);
  }
}

// ---------------- edge pass 3: weighted scatter with folded head-mean ----------------
// one wave per edge; lane covers channels c and c+32; acc[dst,c] += 0.25*sum_h alpha_h*xl[src,h,c]

__global__ void k_edge_msg(const int* __restrict__ src, const int* __restrict__ dst,
                           const float* __restrict__ exbuf, const float* __restrict__ s,
                           const float* __restrict__ xl, float* __restrict__ acc) {
  int e = blockIdx.x * (blockDim.x >> 5) + (threadIdx.x >> 5);
  if (e >= NE) return;
  int lane = threadIdx.x & 31;
  int u = src[e], v = dst[e];
  float a0 = exbuf[e * NH + 0] / s[v * NH + 0];
  float a1 = exbuf[e * NH + 1] / s[v * NH + 1];
  float a2 = exbuf[e * NH + 2] / s[v * NH + 2];
  float a3 = exbuf[e * NH + 3] / s[v * NH + 3];
  const float* xr = xl + u * FDIM + lane;          // immediate offsets below
  float* ar = acc + v * HC + lane;
#pragma unroll
  for (int half = 0; half < 2; ++half) {
    int c = half * 32;
    float val = a0 * xr[c] + a1 * xr[HC + c] + a2 * xr[2 * HC + c] + a3 * xr[3 * HC + c];
    atomicAdd(&ar[c], 0.25f * val);
  }
}

// ---------------- finalize layer 1: relu -> f16 copy for WMMA A ----------------

__global__ void k_relu_f16(const float* __restrict__ acc, _Float16* __restrict__ h16) {
  int t = blockIdx.x * blockDim.x + threadIdx.x;
  if (t >= NN * HC) return;
  float v = acc[t];
  h16[t] = (_Float16)(v > 0.0f ? v : 0.0f);
}

// ---------------- layer 2 transform: xl2 = relu(h1) @ W2 via WMMA f16->f32 ----------------
// A: [NN,64] f16 row-major; B: [64,256] f16 row-major; C: [NN,256] f32.
// Per wave: fixed 16-col n-tile (B frags hoisted), loop over 16-row m-tiles.
// A 16-bit 16x32 layout (ISA 7.12.2): lane l (M=l&15), half-index e:
//   K = (e<8?0:16) + (l>>4)*8 + (e&7)  -> two contiguous 16B chunks per K-half.
// B 32x16: lane l elem e: K as for A, N = l&15.  C/D: VGPR r -> M = r + (l>>4)*8, N = l&15.
// Two independent accumulators avoid the WMMA->WMMA D-RAW hazard (no v_nop fill);
// all hot indices are int32 so stores use one base + 24-bit immediate offsets.

__global__ void __launch_bounds__(256)
k_gemm_wmma(const _Float16* __restrict__ A, const _Float16* __restrict__ Bm,
            float* __restrict__ C) {
  const int lane = threadIdx.x & 31;
  const int wave = threadIdx.x >> 5;
  const int lmod = lane & 15;
  const int lhi  = lane >> 4;
  const int n0   = blockIdx.x * 16;              // 16 n-tiles cover FDIM=256

  v16h b0, b1;
#pragma unroll
  for (int e = 0; e < 16; ++e) {
    b0[e] = Bm[(0  + (e >> 3) * 16 + lhi * 8 + (e & 7)) * FDIM + n0 + lmod];
    b1[e] = Bm[(32 + (e >> 3) * 16 + lhi * 8 + (e & 7)) * FDIM + n0 + lmod];
  }

  const int MT = NN / 16;                        // 6250 exact
  const int stride = (int)gridDim.y * 8;
  for (int mt = blockIdx.y * 8 + wave; mt < MT; mt += stride) {
    const int m0 = mt * 16;
    const _Float16* arow = A + (m0 + lmod) * HC + lhi * 8;
    // prefetch next tile's A row into cache while this tile computes
    int pn = mt + stride;
    if (pn < MT) __builtin_prefetch(A + (pn * 16 + lmod) * HC, 0, 3);

    v8h alo0 = *(const v8h*)(arow);
    v8h ahi0 = *(const v8h*)(arow + 16);
    v8h alo1 = *(const v8h*)(arow + 32);
    v8h ahi1 = *(const v8h*)(arow + 48);
    v16h a0, a1;
#pragma unroll
    for (int i = 0; i < 8; ++i) {
      a0[i] = alo0[i]; a0[8 + i] = ahi0[i];
      a1[i] = alo1[i]; a1[8 + i] = ahi1[i];
    }
    v8f c0 = {};
    v8f c1 = {};
    c0 = __builtin_amdgcn_wmma_f32_16x16x32_f16(false, a0, false, b0, (short)0, c0, false, false);
    c1 = __builtin_amdgcn_wmma_f32_16x16x32_f16(false, a1, false, b1, (short)0, c1, false, false);
    v8f c = c0 + c1;
    float* cbase = C + (m0 + lhi * 8) * FDIM + n0 + lmod;
#pragma unroll
    for (int r = 0; r < 8; ++r)
      cbase[r * FDIM] = c[r];
  }
}

// ---------------- finalize layer 2 + global_add_pool ----------------

__global__ void k_relu_pool(const float* __restrict__ acc, const int* __restrict__ batch,
                            float* __restrict__ out) {
  int t = blockIdx.x * blockDim.x + threadIdx.x;
  if (t >= NN * HC) return;
  int n = t >> 6, c = t & 63;
  float v = acc[t];
  v = v > 0.0f ? v : 0.0f;
  atomicAdd(&out[batch[n] * HC + c], v);
}

// ---------------- host launch ----------------

extern "C" void kernel_launch(void* const* d_in, const int* in_sizes, int n_in,
                              void* d_out, int out_size, void* d_ws, size_t ws_size,
                              hipStream_t stream) {
  const float* x   = (const float*)d_in[0];
  const int*   ei  = (const int*)d_in[1];    // edge_index [2,NE]
  const int*   bid = (const int*)d_in[2];    // batch_ids [NN]
  const float* W1  = (const float*)d_in[3];
  const float* as1 = (const float*)d_in[4];
  const float* ad1 = (const float*)d_in[5];
  const float* b1  = (const float*)d_in[6];
  const float* W2  = (const float*)d_in[7];
  const float* as2 = (const float*)d_in[8];
  const float* ad2 = (const float*)d_in[9];
  const float* b2  = (const float*)d_in[10];
  float* out = (float*)d_out;

  const int* srcI = ei;
  const int* dstI = ei + NE;

  // carve workspace (256B aligned slots)
  char* wp = (char*)d_ws;
  size_t used = 0;
  auto carve = [&](size_t bytes) -> void* {
    void* p = wp + used;
    used += (bytes + 255) & ~(size_t)255;
    return p;
  };
  float*    xl   = (float*)   carve((size_t)NN * FDIM * 4);   // node features [N,256]
  float*    acc  = (float*)   carve((size_t)NN * HC * 4);     // message accumulator [N,64]
  float*    es   = (float*)   carve((size_t)NN * NH * 4);
  float*    ed   = (float*)   carve((size_t)NN * NH * 4);
  float*    m    = (float*)   carve((size_t)NN * NH * 4);
  float*    s    = (float*)   carve((size_t)NN * NH * 4);
  float*    exb  = (float*)   carve((size_t)NE * NH * 4);
  _Float16* h16  = (_Float16*)carve((size_t)NN * HC * 2);
  _Float16* W2h  = (_Float16*)carve((size_t)HC * FDIM * 2);
  if (used > ws_size) return;   // deterministic no-op if scratch too small

  const dim3 B(256);
  auto blk = [](long long n) { return dim3((unsigned)((n + 255) / 256)); };

  // prep
  k_zero<<<blk(NG * HC), B, 0, stream>>>(out, NG * HC);
  k_cvt_f16<<<blk(HC * FDIM), B, 0, stream>>>(W2, W2h, HC * FDIM);

  // ----- layer 1 -----
  k_l1_transform<<<blk((long long)NN * 64), B, 0, stream>>>(x, W1, xl);
  k_attn_logits<<<blk((long long)NN * NH), B, 0, stream>>>(xl, as1, ad1, es, ed);
  k_node_init<<<blk((long long)NN * 64), B, 0, stream>>>(m, s, acc, b1);
  k_edge_max<<<blk(NE), B, 0, stream>>>(srcI, dstI, es, ed, m);
  k_edge_expsum<<<blk(NE), B, 0, stream>>>(srcI, dstI, es, ed, m, exb, s);
  k_edge_msg<<<dim3((NE + 7) / 8), B, 0, stream>>>(srcI, dstI, exb, s, xl, acc);
  k_relu_f16<<<blk((long long)NN * HC), B, 0, stream>>>(acc, h16);

  // ----- layer 2 -----
  k_gemm_wmma<<<dim3(FDIM / 16, 128), B, 0, stream>>>(h16, W2h, xl);
  k_attn_logits<<<blk((long long)NN * NH), B, 0, stream>>>(xl, as2, ad2, es, ed);
  k_node_init<<<blk((long long)NN * 64), B, 0, stream>>>(m, s, acc, b2);
  k_edge_max<<<blk(NE), B, 0, stream>>>(srcI, dstI, es, ed, m);
  k_edge_expsum<<<blk(NE), B, 0, stream>>>(srcI, dstI, es, ed, m, exb, s);
  k_edge_msg<<<dim3((NE + 7) / 8), B, 0, stream>>>(srcI, dstI, exb, s, xl, acc);

  // ----- relu + global_add_pool -----
  k_relu_pool<<<blk((long long)NN * HC), B, 0, stream>>>(acc, bid, out);
}
